// Attention_28724741275707
// MI455X (gfx1250) — matrix operand order repaired and checked by reading, not compiled, not run on hardware
//
#include <hip/hip_runtime.h>
#include <hip/hip_bf16.h>

typedef __attribute__((ext_vector_type(16))) _Float16 v16h;
typedef __attribute__((ext_vector_type(8)))  float    v8f;

constexpr int B_ = 2;
constexpr int S_ = 2048;
constexpr int D_ = 768;
constexpr int H_ = 12;
constexpr int M_ = 64;          // d_head
constexpr float IGNORE_ = -100000.0f;

#define WMMA_F16(A, Bm, C) \
  __builtin_amdgcn_wmma_f32_16x16x32_f16(false, (A), false, (Bm), (short)0, (C), false, false)

#define USE_ASYNC_LDS 1

// ---------------------------------------------------------------------------
// CDNA5 async copy: 16 bytes/lane, global -> LDS, tracked by ASYNCcnt.
// GV addressing form: global_load_async_to_lds_b128 vdst(lds), vaddr64, off
// ---------------------------------------------------------------------------
static __device__ __forceinline__ void async_copy16(_Float16* lds_dst, const _Float16* gsrc) {
#if USE_ASYNC_LDS
  const unsigned loff = (unsigned)(uintptr_t)(void*)lds_dst;  // LDS aperture: low 32 bits = LDS offset
  asm volatile("global_load_async_to_lds_b128 %0, %1, off"
               :: "v"(loff), "v"(gsrc)
               : "memory");
#else
  *(uint4*)lds_dst = *(const uint4*)gsrc;
#endif
}

static __device__ __forceinline__ void wait_async_le(const int n) {
#if USE_ASYNC_LDS
  if (n == 0) asm volatile("s_wait_asynccnt 0x0" ::: "memory");
  else        asm volatile("s_wait_asynccnt 0x2" ::: "memory");
#endif
}

// ---------------------------------------------------------------------------
// Fragment loaders (wave32; A/B k order contiguous, not ISA table order; C/D per CDNA5 ISA 7.12.2)
// A 16x32 f16: lane m = tid&15, hi = tid bit4; pair v: K = hi*16 + v*2
// B 32x16:    lane n = tid&15; element e: K = hi*16 + e
// C/D 16x16 f32: lane col = tid&15; VGPR r: row = hi*8 + r
// ---------------------------------------------------------------------------
static __device__ __forceinline__ v16h load_a_f32(const float* __restrict__ src, int ld) {
  const int m  = threadIdx.x & 15;
  const int hi = (threadIdx.x >> 4) & 1;
  v16h a;
#pragma unroll
  for (int v = 0; v < 8; ++v) {
    const int k = hi * 16 + (v << 1);
    const float2 p = *(const float2*)(src + (size_t)m * ld + k);
    a[2 * v]     = (_Float16)p.x;
    a[2 * v + 1] = (_Float16)p.y;
  }
  return a;
}

static __device__ __forceinline__ v16h load_a_f16(const _Float16* __restrict__ src, int ld) {
  const int m  = threadIdx.x & 15;
  const int hi = (threadIdx.x >> 4) & 1;
  v16h a;
#pragma unroll
  for (int v = 0; v < 8; ++v) {
    const int k = hi * 16 + (v << 1);
    union { unsigned u; _Float16 h[2]; } cv;
    cv.u = *(const unsigned*)(src + (size_t)m * ld + k);
    a[2 * v]     = cv.h[0];
    a[2 * v + 1] = cv.h[1];
  }
  return a;
}

// B fragment from f16 buffer laid out [n][k] (contraction contiguous, ld = row pitch)
static __device__ __forceinline__ v16h load_b_f16_cont(const _Float16* src, int ld) {
  const int n  = threadIdx.x & 15;
  const int hi = (threadIdx.x >> 4) & 1;
  return *(const v16h*)(src + (size_t)n * ld + hi * 16);
}

// ---------------------------------------------------------------------------
// Kernel 1: QKV projection. 8 waves/block = 8 s-tiles, one (b,h) per block.
// W_{Q,K,V} 32xk-chunk staged once per block into LDS as f16 transposed [n][k].
// Writes Q,K row-major [B,H,S,64] f16 and V transposed [B,H,64,S] f16.
// ---------------------------------------------------------------------------
__global__ __launch_bounds__(256) void qkv_proj_kernel(
    const float* __restrict__ x,
    const float* __restrict__ Wq, const float* __restrict__ bq,
    const float* __restrict__ Wk, const float* __restrict__ bk,
    const float* __restrict__ Wv, const float* __restrict__ bv,
    _Float16* __restrict__ Q, _Float16* __restrict__ K, _Float16* __restrict__ Vt) {
  __shared__ _Float16 WT[3][64 * 32];   // [q/k/v][n][kk], 12 KB

  const int w  = threadIdx.x >> 5;
  const int lo = threadIdx.x & 15;
  const int hi = (threadIdx.x >> 4) & 1;
  const int s0 = blockIdx.x * 128 + w * 16;
  const int h  = blockIdx.y;
  const int b  = blockIdx.z;

  const float* xrow = x + ((size_t)b * S_ + s0) * D_;
  const float* wsrc[3] = {Wq + (size_t)h * D_ * M_,
                          Wk + (size_t)h * D_ * M_,
                          Wv + (size_t)h * D_ * M_};

  v8f acc[3][4] = {};

  const int t  = threadIdx.x;
  const int kk = t >> 3;          // 0..31  (k within chunk)
  const int c8 = (t & 7) * 8;     // 8-column group

  for (int kc = 0; kc < D_; kc += 32) {
    __syncthreads();              // previous chunk's reads done before overwrite
#pragma unroll
    for (int wi = 0; wi < 3; ++wi) {
      const float4 f0 = *(const float4*)(wsrc[wi] + (size_t)(kc + kk) * M_ + c8);
      const float4 f1 = *(const float4*)(wsrc[wi] + (size_t)(kc + kk) * M_ + c8 + 4);
      _Float16* dst = &WT[wi][0];
      dst[(c8 + 0) * 32 + kk] = (_Float16)f0.x;
      dst[(c8 + 1) * 32 + kk] = (_Float16)f0.y;
      dst[(c8 + 2) * 32 + kk] = (_Float16)f0.z;
      dst[(c8 + 3) * 32 + kk] = (_Float16)f0.w;
      dst[(c8 + 4) * 32 + kk] = (_Float16)f1.x;
      dst[(c8 + 5) * 32 + kk] = (_Float16)f1.y;
      dst[(c8 + 6) * 32 + kk] = (_Float16)f1.z;
      dst[(c8 + 7) * 32 + kk] = (_Float16)f1.w;
    }
    __syncthreads();

    const v16h a = load_a_f32(xrow + kc, D_);
    // Batch 4 B-fragment DS loads per weight into distinct regs so the LDS
    // pipeline keeps loads in flight across the WMMA issues (amortized waits).
#pragma unroll
    for (int wi = 0; wi < 3; ++wi) {
      v16h bf[4];
#pragma unroll
      for (int nt = 0; nt < 4; ++nt)
        bf[nt] = *(const v16h*)(&WT[wi][(nt * 16 + lo) * 32 + hi * 16]);
#pragma unroll
      for (int nt = 0; nt < 4; ++nt)
        acc[wi][nt] = WMMA_F16(a, bf[nt], acc[wi][nt]);
    }
  }

  const size_t bh = (size_t)b * H_ + h;
#pragma unroll
  for (int nt = 0; nt < 4; ++nt) {
    const int col = nt * 16 + lo;
    const float biq = bq[h * M_ + col];
    const float bik = bk[h * M_ + col];
    const float biv = bv[h * M_ + col];
#pragma unroll
    for (int r = 0; r < 8; ++r) {
      const int row = hi * 8 + r;
      const size_t idx = (bh * S_ + s0 + row) * M_ + col;
      Q[idx] = (_Float16)(acc[0][nt][r] + biq);
      K[idx] = (_Float16)(acc[1][nt][r] + bik);
      Vt[(bh * M_ + col) * S_ + s0 + row] = (_Float16)(acc[2][nt][r] + biv);
    }
  }
}

// ---------------------------------------------------------------------------
// Kernel 2: causal flash attention. 8 waves/block = 8 consecutive q-tiles
// sharing double-buffered K (32x64) and V^T (64x32) f16 tiles staged with
// CDNA5 async-to-LDS (1 x b128 per lane per tile). Online softmax per wave;
// P staged through this wave's LDS slice (DS same-wave in-order).
// ---------------------------------------------------------------------------
__global__ __launch_bounds__(256) void flash_attn_kernel(
    const _Float16* __restrict__ Q, const _Float16* __restrict__ K,
    const _Float16* __restrict__ Vt, _Float16* __restrict__ Z) {
  __shared__ _Float16 KT[2][32 * 64];   // [kpos][d]   8 KB
  __shared__ _Float16 VT[2][64 * 32];   // [d][kpos]   8 KB
  __shared__ _Float16 PT[8][16 * 32];   // per-wave P  8 KB

  const int w    = threadIdx.x >> 5;
  const int lo   = threadIdx.x & 15;
  const int hi   = (threadIdx.x >> 4) & 1;
  const int base = blockIdx.x * 128;
  const int s0   = base + w * 16;
  const int h    = blockIdx.y;
  const int b    = blockIdx.z;

  const size_t bh = (size_t)b * H_ + h;
  const _Float16* Qh = Q  + (bh * S_ + s0) * M_;
  const _Float16* Kh = K  + bh * S_ * M_;
  const _Float16* Vh = Vt + bh * (size_t)M_ * S_;
  _Float16* Pw = &PT[w][0];

  const v16h qa0 = load_a_f16(Qh, M_);       // d = 0..31
  const v16h qa1 = load_a_f16(Qh + 32, M_);  // d = 32..63

  v8f acc[4] = {};
  float mrow[8], lrow[8];
#pragma unroll
  for (int r = 0; r < 8; ++r) { mrow[r] = -3.0e38f; lrow[r] = 0.0f; }

  const int nkb_w   = (s0 + 16 + 31) >> 5;   // this wave's causal k-block limit
  const int nkb_blk = (base + 128) >> 5;     // block-wide staging trip count

  const int t    = threadIdx.x;
  const int krow = t >> 3, kseg = (t & 7) * 8;   // K tile: 32 rows x 64 halves
  const int vrow = t >> 2, vseg = (t & 3) * 8;   // V tile: 64 rows x 32 halves

  auto stage = [&](int kb, int bufi) {
    const int k0 = kb * 32;
    async_copy16(&KT[bufi][krow * 64 + kseg], Kh + (size_t)(k0 + krow) * M_ + kseg);
    async_copy16(&VT[bufi][vrow * 32 + vseg], Vh + (size_t)vrow * S_ + k0 + vseg);
  };

  stage(0, 0);
  for (int kb = 0; kb < nkb_blk; ++kb) {
    const int buf = kb & 1;
    if (kb + 1 < nkb_blk) { stage(kb + 1, buf ^ 1); wait_async_le(2); }
    else                  { wait_async_le(0); }
    __syncthreads();

    if (kb < nkb_w) {                        // wave-uniform: EXEC stays all-1s
      const int k0 = kb * 32;
      const _Float16* Kb = &KT[buf][0];
      const _Float16* Vb = &VT[buf][0];

      // ---- scores: two 16x16 tiles; batch the 4 K fragments first ----
      v16h kf[4];
#pragma unroll
      for (int t2 = 0; t2 < 2; ++t2) {
        kf[2 * t2 + 0] = *(const v16h*)(Kb + (size_t)(t2 * 16 + lo) * 64 + hi * 16);
        kf[2 * t2 + 1] = *(const v16h*)(Kb + (size_t)(t2 * 16 + lo) * 64 + 32 + hi * 16);
      }
      v8f sc[2];
#pragma unroll
      for (int t2 = 0; t2 < 2; ++t2) {
        v8f c = {};
        c = WMMA_F16(qa0, kf[2 * t2 + 0], c);
        c = WMMA_F16(qa1, kf[2 * t2 + 1], c);
        sc[t2] = c;
      }

      // ---- scale + causal mask + online softmax ----
      float newm[8];
#pragma unroll
      for (int r = 0; r < 8; ++r) {
        const int rowg = s0 + hi * 8 + r;
#pragma unroll
        for (int t2 = 0; t2 < 2; ++t2) {
          const int colg = k0 + t2 * 16 + lo;
          const float sv = sc[t2][r] * 0.125f;           // 1/sqrt(64)
          sc[t2][r] = (colg <= rowg) ? sv : IGNORE_;     // reference semantics
        }
        float mx = fmaxf(sc[0][r], sc[1][r]);
        mx = fmaxf(mx, __shfl_xor(mx, 1, 32));
        mx = fmaxf(mx, __shfl_xor(mx, 2, 32));
        mx = fmaxf(mx, __shfl_xor(mx, 4, 32));
        mx = fmaxf(mx, __shfl_xor(mx, 8, 32));
        newm[r] = fmaxf(mrow[r], mx);
      }

#pragma unroll
      for (int r = 0; r < 8; ++r) {
        const float p0 = __expf(sc[0][r] - newm[r]);
        const float p1 = __expf(sc[1][r] - newm[r]);
        float rs = p0 + p1;
        rs += __shfl_xor(rs, 1, 32);
        rs += __shfl_xor(rs, 2, 32);
        rs += __shfl_xor(rs, 4, 32);
        rs += __shfl_xor(rs, 8, 32);
        const float corr = __expf(mrow[r] - newm[r]);
        lrow[r] = lrow[r] * corr + rs;
        mrow[r] = newm[r];
#pragma unroll
        for (int nt = 0; nt < 4; ++nt) acc[nt][r] = acc[nt][r] * corr;
        const int row = hi * 8 + r;
        Pw[row * 32 + lo]      = (_Float16)p0;
        Pw[row * 32 + 16 + lo] = (_Float16)p1;
      }
      // same-wave DS ordering guarantees store->load visibility (no barrier)
      const v16h pa = load_a_f16(Pw, 32);
      v16h vf[4];
#pragma unroll
      for (int nt = 0; nt < 4; ++nt)
        vf[nt] = *(const v16h*)(Vb + (size_t)(nt * 16 + lo) * 32 + hi * 16);
#pragma unroll
      for (int nt = 0; nt < 4; ++nt)
        acc[nt] = WMMA_F16(pa, vf[nt], acc[nt]);
    }
    __syncthreads();
  }

  // ---- epilogue: z = acc / l, stored f16 as [B*S, H*64] ----
  _Float16* Zrow = Z + ((size_t)b * S_ + s0) * (H_ * M_) + h * M_;
#pragma unroll
  for (int r = 0; r < 8; ++r) {
    const float inv = 1.0f / lrow[r];
    const int row = hi * 8 + r;
#pragma unroll
    for (int nt = 0; nt < 4; ++nt)
      Zrow[(size_t)row * (H_ * M_) + nt * 16 + lo] = (_Float16)(acc[nt][r] * inv);
  }
}

// ---------------------------------------------------------------------------
// Kernel 3: output projection. 32 rows x 768 cols per block; the full 32x768
// W_O chunk is staged once per block as f16 transposed [col][k] (48 KB LDS),
// shared by 8 waves (96 cols each); each B fragment reused for 2 row tiles.
// ---------------------------------------------------------------------------
__global__ __launch_bounds__(256) void out_proj_kernel(
    const _Float16* __restrict__ Zin, const float* __restrict__ Wo,
    const float* __restrict__ bo, float* __restrict__ out) {
  __shared__ _Float16 WoT[768 * 32];    // 48 KB

  const int w  = threadIdx.x >> 5;
  const int lo = threadIdx.x & 15;
  const int hi = (threadIdx.x >> 4) & 1;
  const int r0 = blockIdx.x * 32;
  const int c0 = w * 96;

  v8f acc[2][6] = {};
  const _Float16* Zr = Zin + (size_t)r0 * D_;

  const int t  = threadIdx.x;
  const int kk = t >> 3;          // 0..31
  const int cb = (t & 7) * 96;    // 96-column group per thread

  for (int kc = 0; kc < D_; kc += 32) {
    __syncthreads();
#pragma unroll
    for (int j4 = 0; j4 < 24; ++j4) {
      const float4 f = *(const float4*)(Wo + (size_t)(kc + kk) * D_ + cb + j4 * 4);
      WoT[(cb + j4 * 4 + 0) * 32 + kk] = (_Float16)f.x;
      WoT[(cb + j4 * 4 + 1) * 32 + kk] = (_Float16)f.y;
      WoT[(cb + j4 * 4 + 2) * 32 + kk] = (_Float16)f.z;
      WoT[(cb + j4 * 4 + 3) * 32 + kk] = (_Float16)f.w;
    }
    __syncthreads();

    const v16h a0 = load_a_f16(Zr + kc, D_);
    const v16h a1 = load_a_f16(Zr + (size_t)16 * D_ + kc, D_);
    v16h bf[6];
#pragma unroll
    for (int nt = 0; nt < 6; ++nt)
      bf[nt] = *(const v16h*)(&WoT[(c0 + nt * 16 + lo) * 32 + hi * 16]);
#pragma unroll
    for (int nt = 0; nt < 6; ++nt) {
      acc[0][nt] = WMMA_F16(a0, bf[nt], acc[0][nt]);
      acc[1][nt] = WMMA_F16(a1, bf[nt], acc[1][nt]);
    }
  }

#pragma unroll
  for (int half = 0; half < 2; ++half) {
#pragma unroll
    for (int nt = 0; nt < 6; ++nt) {
      const int col  = c0 + nt * 16 + lo;
      const float bi = bo[col];
#pragma unroll
      for (int r = 0; r < 8; ++r) {
        const int row = r0 + half * 16 + hi * 8 + r;
        out[(size_t)row * D_ + col] = acc[half][nt][r] + bi;
      }
    }
  }
}

// ---------------------------------------------------------------------------
extern "C" void kernel_launch(void* const* d_in, const int* in_sizes, int n_in,
                              void* d_out, int out_size, void* d_ws, size_t ws_size,
                              hipStream_t stream) {
  const float* x  = (const float*)d_in[0];
  const float* Wq = (const float*)d_in[1];
  const float* bq = (const float*)d_in[2];
  const float* Wk = (const float*)d_in[3];
  const float* bk = (const float*)d_in[4];
  const float* Wv = (const float*)d_in[5];
  const float* bv = (const float*)d_in[6];
  const float* Wo = (const float*)d_in[7];
  const float* bo = (const float*)d_in[8];
  float* out = (float*)d_out;

  _Float16* ws = (_Float16*)d_ws;
  const size_t qsz = (size_t)B_ * H_ * S_ * M_;  // == B*S*768
  _Float16* Q  = ws;
  _Float16* K  = ws + qsz;
  _Float16* Vt = ws + 2 * qsz;
  _Float16* Z  = ws + 3 * qsz;

  dim3 g12(S_ / 128, H_, B_);
  qkv_proj_kernel<<<g12, 256, 0, stream>>>(x, Wq, bq, Wk, bk, Wv, bv, Q, K, Vt);
  flash_attn_kernel<<<g12, 256, 0, stream>>>(Q, K, Vt, Z);
  out_proj_kernel<<<dim3((B_ * S_) / 32), 256, 0, stream>>>(Z, Wo, bo, out);
}
